// DeepSetModel_2087354105980
// MI455X (gfx1250) — matrix operand order repaired and checked by reading, not compiled, not run on hardware
//
#include <hip/hip_runtime.h>

// ---------------------------------------------------------------------------
// Types / helpers
// ---------------------------------------------------------------------------
typedef __attribute__((ext_vector_type(16))) __bf16 v16bf;
typedef __attribute__((ext_vector_type(8)))  __bf16 v8bf;
typedef __attribute__((ext_vector_type(8)))  float  v8f;
typedef __attribute__((ext_vector_type(4)))  unsigned int v4u;
typedef __attribute__((ext_vector_type(8)))  int v8i;
typedef __attribute__((ext_vector_type(4)))  int v4i;

static __device__ __forceinline__ unsigned short f2bf_bits(float f) {
    unsigned u = __builtin_bit_cast(unsigned, f);
    u += 0x7FFFu + ((u >> 16) & 1u);                 // round-to-nearest-even
    return (unsigned short)(u >> 16);
}
static __device__ __forceinline__ __bf16 f2bf(float f) {
    unsigned short h = f2bf_bits(f);
    return __builtin_bit_cast(__bf16, h);
}
static __device__ __forceinline__ v8f v8f_zero() {
    v8f z;
#pragma unroll
    for (int i = 0; i < 8; ++i) z[i] = 0.0f;
    return z;
}
// A fragment (16x32 bf16, row-major LDS): lane M = lane&15, half = lane>>4.
// elems 0..7  -> K = kk + 8*half .. +7 ; elems 8..15 -> K = kk + 16+8*half .. +7
static __device__ __forceinline__ v16bf load_afrag(const __bf16* row, int kk, int half) {
    union { v16bf v; v8bf h[2]; } u;
    u.h[0] = *(const v8bf*)(row + kk + 8 * half);
    u.h[1] = *(const v8bf*)(row + kk + 16 + 8 * half);
    return u.v;
}
// B fragment (32x16 bf16) from transposed weights wT[n][k]:
// lane N = lane&15, elems 0..7 -> K = kk + 8*half .. +7 ; elems 8..15 -> K = kk + 16+8*half .. +7
static __device__ __forceinline__ v16bf load_bfrag(const __bf16* row, int kk, int half) {
    union { v16bf v; v8bf h[2]; } u;
    u.h[0] = *(const v8bf*)(row + kk + 8 * half);
    u.h[1] = *(const v8bf*)(row + kk + 16 + 8 * half);
    return u.v;
}
static __device__ __forceinline__ void atomic_add_f32(float* p, float v) {
    __hip_atomic_fetch_add(p, v, __ATOMIC_RELAXED, __HIP_MEMORY_SCOPE_AGENT);
}

// ---------------------------------------------------------------------------
// Tensor Data Mover: 1-D global->LDS DMA of nbytes (multiple of 8, /8 <= 65535)
// Hand-packed D# per ISA 8.3/8.4: group0 = {count=1, lds_addr, global_addr, type=2},
// group1 = {data_size=8B, tensor_dim0=n8, tensor_dim1=1, tile_dim0=n8, stride=n8}.
// This toolchain exposes the 6-arg builtin: (v4u, v8i, v4i, v4i, v8i, i32 cpol).
// ---------------------------------------------------------------------------
static __device__ __forceinline__ void tdm_load_1d(unsigned lds_off, const void* gptr,
                                                   unsigned nbytes) {
    unsigned long long ga = (unsigned long long)gptr;
    unsigned n8 = nbytes >> 3;
    v4u g0;
    g0[0] = 1u;                                                  // count=1 (user)
    g0[1] = lds_off;                                             // lds_addr
    g0[2] = (unsigned)(ga & 0xFFFFFFFFu);                        // global_addr[31:0]
    g0[3] = (unsigned)((ga >> 32) & 0x01FFFFFFu) | 0x80000000u;  // [56:32] | type=2
    v8i g1;
    g1[0] = (int)(3u << 16);                         // wg_mask=0, data_size=3 (8B)
    g1[1] = (int)((n8 & 0xFFFFu) << 16);             // tensor_dim0[15:0] @ bits 63:48
    g1[2] = (int)((n8 >> 16) | (1u << 16));          // tensor_dim0[31:16] | tensor_dim1 lo = 1
    g1[3] = (int)((n8 & 0xFFFFu) << 16);             // tensor_dim1 hi = 0 | tile_dim0 = n8
    g1[4] = 0;                                       // tile_dim1 = 0 (unused), tile_dim2 = 0
    g1[5] = (int)n8;                                 // tensor_dim0_stride[31:0]
    g1[6] = 0;                                       // stride hi | dim1_stride lo
    g1[7] = 0;
    v4i z4 = {0, 0, 0, 0};
    v8i z8 = {0, 0, 0, 0, 0, 0, 0, 0};
    __builtin_amdgcn_tensor_load_to_lds(g0, g1, z4, z4, z8, 0);
}
static __device__ __forceinline__ unsigned lds_offset_of(const void* p) {
    return (unsigned)(unsigned long long)p;          // low 32 bits = LDS byte offset
}

// ---------------------------------------------------------------------------
// Segment-id compaction: idxs[i] = #{ j in [1..i] : ev[j] != ev[j-1] }
// ---------------------------------------------------------------------------
__global__ __launch_bounds__(256) void seg_count_kernel(const long long* __restrict__ ev,
                                                        int n, int* __restrict__ blockcnt) {
    __shared__ int red[256];
    int base = blockIdx.x * 1024 + threadIdx.x * 4;
    int c = 0;
#pragma unroll
    for (int j = 0; j < 4; ++j) {
        int i = base + j;
        if (i > 0 && i < n) c += (ev[i] != ev[i - 1]) ? 1 : 0;
    }
    red[threadIdx.x] = c;
    __syncthreads();
    for (int s = 128; s > 0; s >>= 1) {
        if (threadIdx.x < s) red[threadIdx.x] += red[threadIdx.x + s];
        __syncthreads();
    }
    if (threadIdx.x == 0) blockcnt[blockIdx.x] = red[0];
}

__global__ __launch_bounds__(1024) void scan_kernel(const int* __restrict__ blockcnt,
                                                    int* __restrict__ blockoff, int nb) {
    __shared__ int sh[1024];
    int t = threadIdx.x;
    sh[t] = (t < nb) ? blockcnt[t] : 0;
    __syncthreads();
    for (int d = 1; d < 1024; d <<= 1) {
        int v = (t >= d) ? sh[t - d] : 0;
        __syncthreads();
        sh[t] += v;
        __syncthreads();
    }
    if (t < nb) blockoff[t] = (t == 0) ? 0 : sh[t - 1];
}

__global__ __launch_bounds__(256) void seg_write_kernel(const long long* __restrict__ ev,
                                                        int n, const int* __restrict__ blockoff,
                                                        int* __restrict__ idxs) {
    __shared__ int sh[256];
    int t = threadIdx.x;
    int base = blockIdx.x * 1024 + t * 4;
    int ch[4];
    int c = 0;
#pragma unroll
    for (int j = 0; j < 4; ++j) {
        int i = base + j;
        int cc = (i > 0 && i < n && ev[i] != ev[i - 1]) ? 1 : 0;
        ch[j] = cc;
        c += cc;
    }
    sh[t] = c;
    __syncthreads();
    int own = c;
    for (int d = 1; d < 256; d <<= 1) {
        int v = (t >= d) ? sh[t - d] : 0;
        __syncthreads();
        sh[t] += v;
        __syncthreads();
    }
    int run = blockoff[blockIdx.x] + (sh[t] - own);   // exclusive prefix within block
#pragma unroll
    for (int j = 0; j < 4; ++j) {
        int i = base + j;
        if (i < n) {
            run += ch[j];
            idxs[i] = run;
        }
    }
}

// Convert f32 [K][N] -> bf16 transposed [N][K] (for direct TDM copy into LDS)
__global__ __launch_bounds__(256) void cvt_bf16_t_kernel(const float* __restrict__ src,
                                                         unsigned short* __restrict__ dst,
                                                         int K, int N) {
    int i = blockIdx.x * 256 + threadIdx.x;
    if (i < K * N) {
        int k = i / N, n2 = i % N;
        dst[n2 * K + k] = f2bf_bits(src[i]);
    }
}

// ---------------------------------------------------------------------------
// phi MLP + segment-sum scatter.
// 128 rows per block, 256 threads = 8 waves (wave32), 16 rows per wave.
// LDS layout (96KB + overlay):
//   [0      ,16384) xs  : x tile     128x64  bf16   (VALU cvt fill)
//   [16384  ,32768) w1t : W1^T       128x64  bf16   (TDM)
//   [32768  ,65536) h1  :            128x128 bf16
//   [65536  ,98304) w2t : W2^T       128x128 bf16   (TDM)
//   h2 overlays [0, 67584): 128x132 f32 (written after last WMMA reads)
// ---------------------------------------------------------------------------
#define PHI_TM 128
#define NFEAT  64
#define NLAT   128
#define H2S    132

__global__ __launch_bounds__(256) void phi_scatter_kernel(
        const float* __restrict__ x,
        const int* __restrict__ segid,
        const unsigned short* __restrict__ w1tg,  // bf16 W1^T [128][64]
        const float* __restrict__ b1,             // [128]
        const unsigned short* __restrict__ w2tg,  // bf16 W2^T [128][128]
        const float* __restrict__ b2,             // [128]
        float* __restrict__ pooled,               // [nseg][128]
        int n) {
    __shared__ __align__(128) char smem[98304];
    __shared__ int segs[PHI_TM];

    __bf16(*xs)[NFEAT]  = (__bf16(*)[NFEAT])(smem);
    __bf16(*w1t)[NFEAT] = (__bf16(*)[NFEAT])(smem + 16384);
    __bf16(*h1)[NLAT]   = (__bf16(*)[NLAT])(smem + 32768);
    __bf16(*w2t)[NLAT]  = (__bf16(*)[NLAT])(smem + 65536);
    float(*h2)[H2S]     = (float(*)[H2S])(smem);

    const int tid  = threadIdx.x;
    const int lane = tid & 31;
    const int wv   = tid >> 5;
    const int half = lane >> 4;
    const int l15  = lane & 15;
    const long long row0 = (long long)blockIdx.x * PHI_TM;

    // TDM: DMA pre-transposed bf16 weights straight into LDS (wave 0 issues).
    if (wv == 0) {
        unsigned base = lds_offset_of(smem);
        tdm_load_1d(base + 16384, w1tg, 16384);
        tdm_load_1d(base + 65536, w2tg, 32768);
    }

    if (tid < PHI_TM) {
        long long r = row0 + tid;
        segs[tid] = (r < n) ? segid[r] : -1;
    }
    // x tile: vectorized float4 loads, packed 4xbf16 (8B) LDS stores
    for (int i = tid * 4; i < PHI_TM * NFEAT; i += 1024) {
        int r = i >> 6, c = i & (NFEAT - 1);
        long long gr = row0 + r;
        float4 v = {0.0f, 0.0f, 0.0f, 0.0f};
        if (gr < n) v = *(const float4*)(x + gr * NFEAT + c);
        unsigned lo = (unsigned)f2bf_bits(v.x) | ((unsigned)f2bf_bits(v.y) << 16);
        unsigned hi = (unsigned)f2bf_bits(v.z) | ((unsigned)f2bf_bits(v.w) << 16);
        uint2 pk = {lo, hi};
        *(uint2*)(&xs[r][c]) = pk;
    }
    if (wv == 0) __builtin_amdgcn_s_wait_tensorcnt(0);
    __syncthreads();

    const int m0 = wv * 16;
    v8f acc[8];
#pragma unroll
    for (int t = 0; t < 8; ++t) acc[t] = v8f_zero();

    // GEMM1: h1 = relu(x @ W1 + b1),  K = 64 -> 2 k-steps
#pragma unroll
    for (int kk = 0; kk < NFEAT; kk += 32) {
        v16bf a = load_afrag(&xs[m0 + l15][0], kk, half);
#pragma unroll
        for (int t = 0; t < 8; ++t) {
            v16bf b = load_bfrag(&w1t[16 * t + l15][0], kk, half);
            acc[t] = __builtin_amdgcn_wmma_f32_16x16x32_bf16(false, a, false, b,
                                                             (short)0, acc[t], false, false);
        }
    }
#pragma unroll
    for (int t = 0; t < 8; ++t) {
        int col = 16 * t + l15;
        float bias = b1[col];
#pragma unroll
        for (int v = 0; v < 8; ++v)
            h1[m0 + v + 8 * half][col] = f2bf(fmaxf(acc[t][v] + bias, 0.0f));
    }
    __syncthreads();

    // GEMM2: h2 = relu(h1 @ W2 + b2),  K = 128 -> 4 k-steps
#pragma unroll
    for (int t = 0; t < 8; ++t) acc[t] = v8f_zero();
#pragma unroll
    for (int kk = 0; kk < NLAT; kk += 32) {
        v16bf a = load_afrag(&h1[m0 + l15][0], kk, half);
#pragma unroll
        for (int t = 0; t < 8; ++t) {
            v16bf b = load_bfrag(&w2t[16 * t + l15][0], kk, half);
            acc[t] = __builtin_amdgcn_wmma_f32_16x16x32_bf16(false, a, false, b,
                                                             (short)0, acc[t], false, false);
        }
    }
    __syncthreads();   // all LDS reads done -> safe to overlay h2
#pragma unroll
    for (int t = 0; t < 8; ++t) {
        int col = 16 * t + l15;
        float bias = b2[col];
#pragma unroll
        for (int v = 0; v < 8; ++v)
            h2[m0 + v + 8 * half][col] = fmaxf(acc[t][v] + bias, 0.0f);
    }
    __syncthreads();

    // Run-compressed segment scatter: rows are sorted by segment id, so reduce
    // equal-id runs in LDS, then one f32 atomic per (run, column).
    {
        int col  = tid & (NLAT - 1);
        int rbeg = (tid >> 7) * 64;
        float s = 0.0f;
        int cur = -1;
        for (int r = rbeg; r < rbeg + 64; ++r) {
            int sg = segs[r];
            if (sg != cur) {
                if (cur >= 0) atomic_add_f32(&pooled[(long long)cur * NLAT + col], s);
                cur = sg;
                s = 0.0f;
            }
            s += h2[r][col];
        }
        if (cur >= 0) atomic_add_f32(&pooled[(long long)cur * NLAT + col], s);
    }
}

// ---------------------------------------------------------------------------
// rho MLP: pooled[128-tile] -> 256 -> 128 -> sigmoid scalar.
// LDS phases (128KB buffer, overlaid):
//   phase1: ps[128][128] bf16 @0 (32KB) + w1t[256][128] bf16 @32768 (TDM, 64KB)
//   phase2: r1[128][256] bf16 @0 (64KB) + w2t[128][256] bf16 @65536 (TDM, 64KB)
//   phase3: r2[128][132] f32  @0 (67.6KB)
// ---------------------------------------------------------------------------
#define RHO_TM 128

__global__ __launch_bounds__(256) void rho_kernel(
        const float* __restrict__ pooled,
        const unsigned short* __restrict__ w1tg,  // bf16 rho_w1^T [256][128]
        const float* __restrict__ b1,             // [256]
        const unsigned short* __restrict__ w2tg,  // bf16 rho_w2^T [128][256]
        const float* __restrict__ b2,             // [128]
        const float* __restrict__ w3,             // [128]
        const float* __restrict__ b3,             // [1]
        float* __restrict__ out,
        int nseg) {
    __shared__ __align__(128) char smem[131072];
    __shared__ float w3s[128];

    __bf16(*ps)[128]  = (__bf16(*)[128])(smem);
    __bf16(*w1t)[128] = (__bf16(*)[128])(smem + 32768);  // [256][128]
    __bf16(*r1)[256]  = (__bf16(*)[256])(smem);
    __bf16(*w2t)[256] = (__bf16(*)[256])(smem + 65536);  // [128][256]
    float(*r2)[H2S]   = (float(*)[H2S])(smem);

    const int tid  = threadIdx.x;
    const int lane = tid & 31;
    const int wv   = tid >> 5;
    const int half = lane >> 4;
    const int l15  = lane & 15;
    const int row0 = blockIdx.x * RHO_TM;

    if (wv == 0) tdm_load_1d(lds_offset_of(smem) + 32768, w1tg, 65536);

    if (tid < 128) w3s[tid] = w3[tid];
    for (int i = tid * 4; i < 128 * 128; i += 1024) {
        int r = i >> 7, c = i & 127;
        int gr = row0 + r;
        float4 v = {0.0f, 0.0f, 0.0f, 0.0f};
        if (gr < nseg) v = *(const float4*)(pooled + (long long)gr * 128 + c);
        unsigned lo = (unsigned)f2bf_bits(v.x) | ((unsigned)f2bf_bits(v.y) << 16);
        unsigned hi = (unsigned)f2bf_bits(v.z) | ((unsigned)f2bf_bits(v.w) << 16);
        uint2 pk = {lo, hi};
        *(uint2*)(&ps[r][c]) = pk;
    }
    if (wv == 0) __builtin_amdgcn_s_wait_tensorcnt(0);
    __syncthreads();

    const int m0 = wv * 16;
    // GEMM1: [128x128] @ [128x256], 16 column tiles, 4 k-steps
    v8f a1[16];
#pragma unroll
    for (int t = 0; t < 16; ++t) a1[t] = v8f_zero();
#pragma unroll
    for (int kk = 0; kk < 128; kk += 32) {
        v16bf a = load_afrag(&ps[m0 + l15][0], kk, half);
#pragma unroll
        for (int t = 0; t < 16; ++t) {
            v16bf b = load_bfrag(&w1t[16 * t + l15][0], kk, half);
            a1[t] = __builtin_amdgcn_wmma_f32_16x16x32_bf16(false, a, false, b,
                                                            (short)0, a1[t], false, false);
        }
    }
    __syncthreads();   // done reading ps/w1t -> overlay with r1/w2t
#pragma unroll
    for (int t = 0; t < 16; ++t) {
        int col = 16 * t + l15;
        float bias = b1[col];
#pragma unroll
        for (int v = 0; v < 8; ++v)
            r1[m0 + v + 8 * half][col] = f2bf(fmaxf(a1[t][v] + bias, 0.0f));
    }
    if (wv == 0) {
        tdm_load_1d(lds_offset_of(smem) + 65536, w2tg, 65536);
        __builtin_amdgcn_s_wait_tensorcnt(0);
    }
    __syncthreads();

    // GEMM2: [128x256] @ [256x128], 8 column tiles, 8 k-steps
    v8f a2[8];
#pragma unroll
    for (int t = 0; t < 8; ++t) a2[t] = v8f_zero();
#pragma unroll
    for (int kk = 0; kk < 256; kk += 32) {
        v16bf a = load_afrag(&r1[m0 + l15][0], kk, half);
#pragma unroll
        for (int t = 0; t < 8; ++t) {
            v16bf b = load_bfrag(&w2t[16 * t + l15][0], kk, half);
            a2[t] = __builtin_amdgcn_wmma_f32_16x16x32_bf16(false, a, false, b,
                                                            (short)0, a2[t], false, false);
        }
    }
    __syncthreads();   // all reads done -> overlay with r2
#pragma unroll
    for (int t = 0; t < 8; ++t) {
        int col = 16 * t + l15;
        float bias = b2[col];
#pragma unroll
        for (int v = 0; v < 8; ++v)
            r2[m0 + v + 8 * half][col] = fmaxf(a2[t][v] + bias, 0.0f);
    }
    __syncthreads();

    // out = sigmoid(r2 @ w3 + b3)
    if (tid < 128) {
        int gr = row0 + tid;
        if (gr < nseg) {
            float s = b3[0];
            for (int c = 0; c < 128; ++c) s += r2[tid][c] * w3s[c];
            out[gr] = 1.0f / (1.0f + __expf(-s));
        }
    }
}

// ---------------------------------------------------------------------------
// Host launcher
// ---------------------------------------------------------------------------
extern "C" void kernel_launch(void* const* d_in, const int* in_sizes, int n_in,
                              void* d_out, int out_size, void* d_ws, size_t ws_size,
                              hipStream_t stream) {
    const float*     x      = (const float*)d_in[0];
    const long long* ev     = (const long long*)d_in[1];   // int64 event ids
    const float*     phi_w1 = (const float*)d_in[2];
    const float*     phi_b1 = (const float*)d_in[3];
    const float*     phi_w2 = (const float*)d_in[4];
    const float*     phi_b2 = (const float*)d_in[5];
    const float*     rho_w1 = (const float*)d_in[6];
    const float*     rho_b1 = (const float*)d_in[7];
    const float*     rho_w2 = (const float*)d_in[8];
    const float*     rho_b2 = (const float*)d_in[9];
    const float*     rho_w3 = (const float*)d_in[10];
    const float*     rho_b3 = (const float*)d_in[11];
    float*           out    = (float*)d_out;

    const int n    = in_sizes[0] / NFEAT;   // 1,000,000
    const int nseg = out_size;              // 10,000
    const int nb   = (n + 1023) / 1024;     // <= 1024 for single-block scan

    // Workspace carve-out (256B aligned sections)
    char*  ws  = (char*)d_ws;
    size_t off = 0;
    auto carve = [&](size_t bytes) -> char* {
        char* p = ws + off;
        off = (off + bytes + 255) & ~(size_t)255;
        return p;
    };
    int*            idxs     = (int*)carve((size_t)n * sizeof(int));
    float*          pooled   = (float*)carve((size_t)nseg * NLAT * sizeof(float));
    int*            blockcnt = (int*)carve((size_t)nb * sizeof(int));
    int*            blockoff = (int*)carve((size_t)nb * sizeof(int));
    unsigned short* w1tg     = (unsigned short*)carve(64 * 128 * 2);    // [128][64]
    unsigned short* w2tg     = (unsigned short*)carve(128 * 128 * 2);   // [128][128]
    unsigned short* rw1tg    = (unsigned short*)carve(128 * 256 * 2);   // [256][128]
    unsigned short* rw2tg    = (unsigned short*)carve(256 * 128 * 2);   // [128][256]

    (void)hipMemsetAsync(pooled, 0, (size_t)nseg * NLAT * sizeof(float), stream);

    seg_count_kernel<<<nb, 256, 0, stream>>>(ev, n, blockcnt);
    scan_kernel<<<1, 1024, 0, stream>>>(blockcnt, blockoff, nb);
    seg_write_kernel<<<nb, 256, 0, stream>>>(ev, n, blockoff, idxs);

    cvt_bf16_t_kernel<<<(64 * 128 + 255) / 256, 256, 0, stream>>>(phi_w1, w1tg, 64, 128);
    cvt_bf16_t_kernel<<<(128 * 128 + 255) / 256, 256, 0, stream>>>(phi_w2, w2tg, 128, 128);
    cvt_bf16_t_kernel<<<(128 * 256 + 255) / 256, 256, 0, stream>>>(rho_w1, rw1tg, 128, 256);
    cvt_bf16_t_kernel<<<(256 * 128 + 255) / 256, 256, 0, stream>>>(rho_w2, rw2tg, 256, 128);

    phi_scatter_kernel<<<(n + PHI_TM - 1) / PHI_TM, 256, 0, stream>>>(
        x, idxs, w1tg, phi_b1, w2tg, phi_b2, pooled, n);

    rho_kernel<<<(nseg + RHO_TM - 1) / RHO_TM, 256, 0, stream>>>(
        pooled, rw1tg, rho_b1, rw2tg, rho_b2, rho_w3, rho_b3, out, nseg);
}
